// CustomScaledDotProductAttention_6210522710207
// MI455X (gfx1250) — compile-verified, matrix operand order checked by reading, not run on hardware
//
#include <hip/hip_runtime.h>
#include <cstdint>
#include <cstddef>

// Problem constants (reference: B=2, H=12, S=2048, D=64, fp32 in/out)
#define B_ 2
#define H_ 12
#define S_ 2048
#define D_ 64
#define BH_ (B_ * H_)

typedef __attribute__((ext_vector_type(16))) __bf16 v16bf;
typedef __attribute__((ext_vector_type(8)))  float  v8f;
typedef __attribute__((ext_vector_type(2)))  float  f32x2;
typedef __attribute__((ext_vector_type(2)))  __bf16 bf16x2;

// fp32 -> bf16 RNE as raw bits (used in cold pre-pass only)
__device__ __forceinline__ uint16_t bf_bits(float f) {
    uint32_t u = __builtin_bit_cast(uint32_t, f);
    u += 0x7FFFu + ((u >> 16) & 1u);
    return (uint16_t)(u >> 16);
}
__device__ __forceinline__ uint32_t bf_pack(float lo, float hi) {
    return (uint32_t)bf_bits(lo) | ((uint32_t)bf_bits(hi) << 16);
}
// hot-path packed convert: let the backend pick v_cvt_pk_bf16_f32
__device__ __forceinline__ uint32_t cvt_pk_bf16(float lo, float hi) {
    f32x2 f; f.x = lo; f.y = hi;
    bf16x2 h = __builtin_convertvector(f, bf16x2);
    return __builtin_bit_cast(uint32_t, h);
}

// K-offset pattern for 16-bit WMMA A-fragments (ISA 7.12.2), assumed mirrored
// for B-fragments: element pair i sits at K = (i/4)*16 + half*8 + (i%4)*2
__device__ __forceinline__ int koff_of(int i, int half) {
    return (i >> 2) * 16 + half * 8 + (i & 3) * 2;
}

// swap values between the two 16-lane halves of the wave (identity selectors)
__device__ __forceinline__ float xhalf_swap(float x) {
    int i = __builtin_bit_cast(int, x);
    int y = __builtin_amdgcn_permlanex16(i, i, 0x76543210, 0xfedcba98, false, false);
    return __builtin_bit_cast(float, y);
}

// =====================================================================
// Pre-pass 1: pack K (fp32 row-major) into bf16 A-fragments of K-tiles.
// Layout: kw[bh][t][f][lane][16]  t = key tile of 16, f = feature chunk of 32
// Fragment semantics: A[m][kdim] = K[key t*16 + m][feat f*32 + kpattern]
// =====================================================================
__global__ void __launch_bounds__(256)
pack_k_kernel(const float* __restrict__ k, uint16_t* __restrict__ kw)
{
    int idx = blockIdx.x * blockDim.x + threadIdx.x;   // (bh, t, f, lane)
    const int lane = idx & 31;  idx >>= 5;
    const int f    = idx & 1;   idx >>= 1;
    const int t    = idx & (S_ / 16 - 1); idx >>= 7;   // S/16 = 128
    const int bh   = idx;
    if (bh >= BH_) return;

    const int half = lane >> 4, lm = lane & 15;
    const float* krow = k + ((size_t)bh * S_ + t * 16 + lm) * D_ + f * 32;

    uint32_t dw[8];
#pragma unroll
    for (int i = 0; i < 8; ++i) {
        float2 x = *(const float2*)(krow + koff_of(i, half));
        dw[i] = bf_pack(x.x, x.y);
    }
    uint32_t* out = (uint32_t*)kw + (size_t)(blockIdx.x * blockDim.x + threadIdx.x) * 8;
    *(uint4*)(out)     = make_uint4(dw[0], dw[1], dw[2], dw[3]);
    *(uint4*)(out + 4) = make_uint4(dw[4], dw[5], dw[6], dw[7]);
}

// =====================================================================
// Pre-pass 2: pack V (fp32 row-major) into bf16 A-fragments of V^T.
// Layout: vw[bh][j][o][lane][16]  j = key tile of 32 (chain), o = d-tile of 16
// Fragment semantics: A[m][kk] = V^T[d o*16+m][key j*32 + kpattern]
//                              = V[key j*32 + kpattern][d o*16 + m]
// =====================================================================
__global__ void __launch_bounds__(256)
pack_v_kernel(const float* __restrict__ v, uint16_t* __restrict__ vw)
{
    int idx = blockIdx.x * blockDim.x + threadIdx.x;   // (bh, j, o, lane)
    const int lane = idx & 31;  idx >>= 5;
    const int o    = idx & 3;   idx >>= 2;
    const int j    = idx & (S_ / 32 - 1); idx >>= 6;   // S/32 = 64
    const int bh   = idx;
    if (bh >= BH_) return;

    const int half = lane >> 4, lm = lane & 15;
    const float* vcol = v + ((size_t)bh * S_ + j * 32) * D_ + o * 16 + lm;

    uint32_t dw[8];
#pragma unroll
    for (int i = 0; i < 8; ++i) {
        const int kk = koff_of(i, half);
        dw[i] = bf_pack(vcol[(size_t)kk * D_], vcol[(size_t)(kk + 1) * D_]);
    }
    uint32_t* out = (uint32_t*)vw + (size_t)(blockIdx.x * blockDim.x + threadIdx.x) * 8;
    *(uint4*)(out)     = make_uint4(dw[0], dw[1], dw[2], dw[3]);
    *(uint4*)(out + 4) = make_uint4(dw[4], dw[5], dw[6], dw[7]);
}

// =====================================================================
// Main flash-attention kernel, TRANSPOSED formulation:
//   S^T = K · Q^T      (keys on M, queries on N)
//   O^T = V^T · P^T    (d on M, queries on N)
// One wave owns 16 queries (N = lane%16); 4 independent waves / block.
// KV consumed in 64-key blocks: 8 score + 8 PV + 2 row-sum WMMAs per block.
// P^T B-fragments are built lane-locally from the score C-fragments:
// no LDS, no cross-lane traffic for the softmax matrix.
// =====================================================================
__global__ void __launch_bounds__(128)
sdpa_fa_wmma(const float* __restrict__ q, const uint16_t* __restrict__ kw,
             const uint16_t* __restrict__ vw, const float* __restrict__ mask,
             float* __restrict__ out)
{
    const int lane = threadIdx.x & 31;
    const int wave = threadIdx.x >> 5;
    const int half = lane >> 4;
    const int lm   = lane & 15;

    const int bh = blockIdx.y;
    const int q0 = blockIdx.x * 64 + wave * 16;
    const int myq = q0 + lm;                  // this lane's query row

    const float* qp = q   + (size_t)bh * S_ * D_;
    float*       op = out + (size_t)bh * S_ * D_;
    const float* mrow = mask + (size_t)myq * S_;   // lane's mask row (hoisted)
    const uint16_t* kwp = kw + ((size_t)bh * (S_ / 16) * 2 * 32 + lane) * 16;
    const uint16_t* vwp = vw + ((size_t)bh * (S_ / 32) * 4 * 32 + lane) * 16;

    int koff[8];
#pragma unroll
    for (int i = 0; i < 8; ++i) koff[i] = koff_of(i, half);

    // ---- Q^T B-fragments: B[d][n] = Q[q0+n][d], same gather as an A-frag ----
    v16bf qa[2];
#pragma unroll
    for (int f = 0; f < 2; ++f) {
        union { uint32_t d[8]; v16bf v; } uq;
        const float* qrow = qp + (size_t)myq * D_ + f * 32;
#pragma unroll
        for (int i = 0; i < 8; ++i) {
            float2 x = *(const float2*)(qrow + koff[i]);
            uq.d[i] = cvt_pk_bf16(x.x, x.y);
        }
        qa[f] = uq.v;
    }

    // ---- all-ones A-fragment for row-sum-via-WMMA ----
    v16bf ones;
#pragma unroll
    for (int i = 0; i < 16; ++i)
        ones[i] = __builtin_bit_cast(__bf16, (uint16_t)0x3F80);

    // ---- state: O^T accumulators (4 d-tiles) + per-query lane-scalar stats ----
    v8f o_accT[4] = {};
    float m_i = -INFINITY, l_i = 0.0f;

    const float scale = 0.125f;  // 1/sqrt(64)

    for (int it = 0; it < S_ / 64; ++it) {
        const int kv0 = it * 64;

        // ---------- scores S^T: four 16(key)x16(query) tiles ----------
        v8f s_t[4];
#pragma unroll
        for (int kt = 0; kt < 4; ++kt) {
            const uint16_t* kf = kwp + (size_t)((it * 4 + kt) * 2) * 32 * 16;
            v16bf kb0 = *(const v16bf*)(kf);
            v16bf kb1 = *(const v16bf*)(kf + 32 * 16);
            v8f acc = {};
            acc = __builtin_amdgcn_wmma_f32_16x16x32_bf16(
                false, kb0, false, qa[0], (short)0, acc, false, false);
            acc = __builtin_amdgcn_wmma_f32_16x16x32_bf16(
                false, kb1, false, qa[1], (short)0, acc, false, false);
            s_t[kt] = acc;
        }

        // ---------- scale + additive mask (keys contiguous per lane) ----------
#pragma unroll
        for (int kt = 0; kt < 4; ++kt) {
            union { float4 v[2]; float a[8]; } mu;
            const float* mp = mrow + kv0 + kt * 16 + 8 * half;
            mu.v[0] = *(const float4*)(mp);
            mu.v[1] = *(const float4*)(mp + 4);
#pragma unroll
            for (int r = 0; r < 8; ++r)
                s_t[kt][r] = s_t[kt][r] * scale + mu.a[r];
        }

        // ---------- online softmax: all 32 scores belong to query lm ----------
        float tm[4];
#pragma unroll
        for (int kt = 0; kt < 4; ++kt) {
            float a0 = fmaxf(s_t[kt][0], s_t[kt][1]);
            float a1 = fmaxf(s_t[kt][2], s_t[kt][3]);
            float a2 = fmaxf(s_t[kt][4], s_t[kt][5]);
            float a3 = fmaxf(s_t[kt][6], s_t[kt][7]);
            tm[kt] = fmaxf(fmaxf(a0, a1), fmaxf(a2, a3));
        }
        float bm = fmaxf(fmaxf(tm[0], tm[1]), fmaxf(tm[2], tm[3]));
        bm = fmaxf(bm, xhalf_swap(bm));            // other 32 keys in partner lane
        const float m_new = fmaxf(m_i, bm);
        const float alpha = __expf(m_i - m_new);
        m_i = m_new;

#pragma unroll
        for (int kt = 0; kt < 4; ++kt)
#pragma unroll
            for (int r = 0; r < 8; ++r)
                s_t[kt][r] = __expf(s_t[kt][r] - m_new);

        // rescale O^T by lane-scalar alpha
#pragma unroll
        for (int o = 0; o < 4; ++o)
#pragma unroll
            for (int r = 0; r < 8; ++r)
                o_accT[o][r] *= alpha;

        // ---------- build P^T B-fragments lane-locally ----------
        // chain c element pair i: key = c*32 + koff(i,half)
        //   -> tile kt = 2c + (i/4), row M = half*8 + (i%4)*2 = this lane's
        //      C rows r=(i%4)*2, r+1.  Pure register repack + packed cvt.
        v16bf pab[2];
#pragma unroll
        for (int c = 0; c < 2; ++c) {
            union { uint32_t d[8]; v16bf v; } pu;
#pragma unroll
            for (int i = 0; i < 8; ++i) {
                const int kt = 2 * c + (i >> 2);
                const int r  = (i & 3) * 2;
                pu.d[i] = cvt_pk_bf16(s_t[kt][r], s_t[kt][r + 1]);
            }
            pab[c] = pu.v;
        }

        // ---------- row sums via WMMA: Ones(16x32) @ P^T -> column sums ----------
        v8f lsum = {};
        lsum = __builtin_amdgcn_wmma_f32_16x16x32_bf16(
            false, ones, false, pab[0], (short)0, lsum, false, false);
        lsum = __builtin_amdgcn_wmma_f32_16x16x32_bf16(
            false, ones, false, pab[1], (short)0, lsum, false, false);
        l_i = l_i * alpha + lsum[0];   // every C row equals the column sum

        // ---------- O^T += V^T (64d x 64k) @ P^T (64k x 16q) ----------
#pragma unroll
        for (int o = 0; o < 4; ++o) {
            const uint16_t* vf0 = vwp + (size_t)(((it * 2 + 0) * 4 + o)) * 32 * 16;
            const uint16_t* vf1 = vwp + (size_t)(((it * 2 + 1) * 4 + o)) * 32 * 16;
            v16bf va0 = *(const v16bf*)vf0;
            v16bf va1 = *(const v16bf*)vf1;
            o_accT[o] = __builtin_amdgcn_wmma_f32_16x16x32_bf16(
                false, va0, false, pab[0], (short)0, o_accT[o], false, false);
            o_accT[o] = __builtin_amdgcn_wmma_f32_16x16x32_bf16(
                false, va1, false, pab[1], (short)0, o_accT[o], false, false);
        }
    }

    // ---------- epilogue: O = (O^T)^T / l, vectorized b128 stores ----------
    const float inv = 1.0f / l_i;
#pragma unroll
    for (int o = 0; o < 4; ++o) {
        float* ob = op + (size_t)myq * D_ + o * 16 + 8 * half;
        float4 lo = make_float4(o_accT[o][0] * inv, o_accT[o][1] * inv,
                                o_accT[o][2] * inv, o_accT[o][3] * inv);
        float4 hi = make_float4(o_accT[o][4] * inv, o_accT[o][5] * inv,
                                o_accT[o][6] * inv, o_accT[o][7] * inv);
        *(float4*)(ob)     = lo;
        *(float4*)(ob + 4) = hi;
    }
}

extern "C" void kernel_launch(void* const* d_in, const int* in_sizes, int n_in,
                              void* d_out, int out_size, void* d_ws, size_t ws_size,
                              hipStream_t stream) {
    (void)in_sizes; (void)n_in; (void)out_size; (void)ws_size;
    const float* q    = (const float*)d_in[0];
    const float* k    = (const float*)d_in[1];
    const float* v    = (const float*)d_in[2];
    const float* mask = (const float*)d_in[3];
    float* out        = (float*)d_out;

    // workspace partition: bf16 K-fragments then bf16 V-fragments (12.6 MB)
    const size_t kw_bytes = (size_t)BH_ * S_ * D_ * sizeof(uint16_t);
    uint16_t* kw = (uint16_t*)d_ws;
    uint16_t* vw = (uint16_t*)((char*)d_ws + kw_bytes);

    const int pack_threads = BH_ * (S_ / 16) * 2 * 32;   // 196608
    pack_k_kernel<<<pack_threads / 256, 256, 0, stream>>>(k, kw);
    pack_v_kernel<<<pack_threads / 256, 256, 0, stream>>>(v, vw);

    dim3 grid(S_ / 64, BH_);   // 32 query tiles x 24 (batch*head)
    dim3 block(128);           // 4 waves; each wave owns 16 queries
    sdpa_fa_wmma<<<grid, block, 0, stream>>>(q, kw, vw, mask, out);
}
